// HierarchicalGraphNetwork_82952998355944
// MI455X (gfx1250) — matrix operand order repaired and checked by reading, not compiled, not run on hardware
//
#include <hip/hip_runtime.h>
#include <cstdint>
#include <cstddef>

// ---------------- problem constants ----------------
#define NN   20000
#define EE   160000
#define NN1  16000
#define NN2  9600
#define HID  64
#define HEADS 8

typedef _Float16 f16_t;
typedef __attribute__((ext_vector_type(16))) _Float16 v16h;
typedef __attribute__((ext_vector_type(8)))  _Float16 v8h;
typedef __attribute__((ext_vector_type(8)))  float    v8f;

static inline int cdiv(long a, long b){ return (int)((a + b - 1) / b); }

__device__ __forceinline__ float lrelu02(float v){ return v > 0.f ? v : 0.2f * v; }
// order-preserving float<->uint mapping for atomicMax on floats (any sign)
__device__ __forceinline__ unsigned f2ord(float f){
    unsigned u = __float_as_uint(f);
    return (u & 0x80000000u) ? ~u : (u | 0x80000000u);
}
__device__ __forceinline__ float ord2f(unsigned u){
    return (u & 0x80000000u) ? __uint_as_float(u & 0x7fffffffu) : __uint_as_float(~u);
}

// ---------------- elementwise / scatter kernels ----------------
__global__ void k_zero(float* __restrict__ p, long n){
    long i = (long)blockIdx.x * blockDim.x + threadIdx.x;
    if (i < n) p[i] = 0.f;
}

__global__ void k_pack_f16(const float* __restrict__ s, f16_t* __restrict__ d, long n){
    long i = (long)blockIdx.x * blockDim.x + threadIdx.x;
    if (i < n) d[i] = (f16_t)s[i];
}

// SAGE: sum_{j->i} x_j and in-degree count
__global__ void k_edge_accum(const float* __restrict__ x, const int* __restrict__ src,
                             const int* __restrict__ dst, float* __restrict__ sum,
                             float* __restrict__ cnt, int C){
    long tid = (long)blockIdx.x * blockDim.x + threadIdx.x;
    long tot = (long)EE * C;
    if (tid >= tot) return;
    int e = (int)(tid / C), c = (int)(tid % C);
    int s = src[e], t = dst[e];
    atomicAdd(&sum[(long)t * C + c], x[(long)s * C + c]);
    if (c == 0) atomicAdd(&cnt[t], 1.f);
}

__global__ void k_mean(float* __restrict__ sum, const float* __restrict__ cnt, int n, int C){
    long tid = (long)blockIdx.x * blockDim.x + threadIdx.x;
    if (tid >= (long)n * C) return;
    int i = (int)(tid / C);
    sum[tid] /= fmaxf(cnt[i], 1.f);
}

__global__ void k_gather(float* __restrict__ d, const float* __restrict__ s,
                         const int* __restrict__ perm, int n, int C){
    long tid = (long)blockIdx.x * blockDim.x + threadIdx.x;
    if (tid >= (long)n * C) return;
    int i = (int)(tid / C), c = (int)(tid % C);
    d[tid] = s[(long)perm[i] * C + c];
}

// GAT: per-node attention logits + seed segment-max with the self-loop value
__global__ void k_att_node(const float* __restrict__ H, const float* __restrict__ att_s,
                           const float* __restrict__ att_d, float* __restrict__ asrc,
                           float* __restrict__ adst, unsigned* __restrict__ nmax,
                           int n, int Hn){
    long tid = (long)blockIdx.x * blockDim.x + threadIdx.x;
    if (tid >= (long)n * Hn) return;
    int i = (int)(tid / Hn), h = (int)(tid % Hn);
    const float* hp = H + (long)i * (Hn * 64) + h * 64;
    float s = 0.f, d = 0.f;
    #pragma unroll 8
    for (int c = 0; c < 64; ++c){ s += hp[c] * att_s[h * 64 + c]; d += hp[c] * att_d[h * 64 + c]; }
    asrc[tid] = s; adst[tid] = d;
    nmax[tid] = f2ord(lrelu02(s + d));   // self-loop always present
}

__global__ void k_edge_max(const int* __restrict__ src, const int* __restrict__ dst,
                           const float* __restrict__ asrc, const float* __restrict__ adst,
                           unsigned* __restrict__ nmax, int Hn, int lim){
    long tid = (long)blockIdx.x * blockDim.x + threadIdx.x;
    if (tid >= (long)EE * Hn) return;
    int e = (int)(tid / Hn), h = (int)(tid % Hn);
    int s = src[e], t = dst[e];
    if (s < lim && t < lim){
        float v = lrelu02(asrc[(long)s * Hn + h] + adst[(long)t * Hn + h]);
        atomicMax(&nmax[(long)t * Hn + h], f2ord(v));
    }
}

// finalize max, init denominator and output with self-loop contribution
__global__ void k_node_init(const float* __restrict__ H, const float* __restrict__ asrc,
                            const float* __restrict__ adst, const unsigned* __restrict__ nmax,
                            float* __restrict__ mfl, float* __restrict__ nden,
                            float* __restrict__ O, int n, int Hn){
    long tid = (long)blockIdx.x * blockDim.x + threadIdx.x;
    if (tid >= (long)n * Hn) return;
    int i = (int)(tid / Hn), h = (int)(tid % Hn);
    float m = ord2f(nmax[tid]);
    mfl[tid] = m;
    float es = lrelu02(asrc[tid] + adst[tid]);
    float ps = __expf(es - m);
    nden[tid] = ps;
    const float* hp = H + (long)i * (Hn * 64) + h * 64;
    float* op = O + (long)i * (Hn * 64) + h * 64;
    #pragma unroll 8
    for (int c = 0; c < 64; ++c) op[c] = ps * hp[c];
}

__global__ void k_edge_soft(const int* __restrict__ src, const int* __restrict__ dst,
                            const float* __restrict__ asrc, const float* __restrict__ adst,
                            const float* __restrict__ mfl, float* __restrict__ nden,
                            float* __restrict__ edge_p, int Hn, int lim){
    long tid = (long)blockIdx.x * blockDim.x + threadIdx.x;
    if (tid >= (long)EE * Hn) return;
    int e = (int)(tid / Hn), h = (int)(tid % Hn);
    int s = src[e], t = dst[e];
    if (s < lim && t < lim){
        float v = lrelu02(asrc[(long)s * Hn + h] + adst[(long)t * Hn + h]);
        float p = __expf(v - mfl[(long)t * Hn + h]);
        edge_p[tid] = p;
        atomicAdd(&nden[(long)t * Hn + h], p);
    }
}

__global__ void k_edge_scatter(const int* __restrict__ src, const int* __restrict__ dst,
                               const float* __restrict__ edge_p, const float* __restrict__ H,
                               float* __restrict__ O, int Hn, int lim){
    long tid = (long)blockIdx.x * blockDim.x + threadIdx.x;
    if (tid >= (long)EE * Hn) return;
    int e = (int)(tid / Hn), h = (int)(tid % Hn);
    int s = src[e], t = dst[e];
    if (s < lim && t < lim){
        float p = edge_p[tid];
        const float* hp = H + (long)s * (Hn * 64) + h * 64;
        float* op = O + (long)t * (Hn * 64) + h * 64;
        #pragma unroll 4
        for (int c = 0; c < 64; ++c) atomicAdd(&op[c], p * hp[c]);
    }
}

// out = out/(den+1e-16) + bias ; optional ELU ; in place
__global__ void k_gat_finish(float* __restrict__ O, const float* __restrict__ nden,
                             const float* __restrict__ bias, int n, int Hn, int do_elu){
    long tid = (long)blockIdx.x * blockDim.x + threadIdx.x;
    int HC = Hn * 64;
    if (tid >= (long)n * HC) return;
    int i = (int)(tid / HC), j = (int)(tid % HC), h = j >> 6;
    float v = O[tid] / (nden[(long)i * Hn + h] + 1e-16f) + bias[j];
    if (do_elu) v = v > 0.f ? v : (__expf(v) - 1.f);
    O[tid] = v;
}

__global__ void k_combine(const float* __restrict__ f0, const float* __restrict__ f1,
                          const float* __restrict__ f2, float* __restrict__ comb){
    long tid = (long)blockIdx.x * blockDim.x + threadIdx.x;
    if (tid >= (long)NN * 192) return;
    int i = (int)(tid / 192), c = (int)(tid % 192);
    float v;
    if (c < 64)        v = f0[(long)i * 64 + c];
    else if (c < 128)  v = (i < NN1) ? f1[(long)i * 64 + (c - 64)]  : 0.f;
    else               v = (i < NN2) ? f2[(long)i * 64 + (c - 128)] : 0.f;
    comb[tid] = v;
}

// ---------------- WMMA GEMM: C[M,N] (+)= A[M,K]f32 @ B[K,N]f16 ----------------
// Block = 4 waves = 64 rows x 64 cols of C. Each wave: 16 rows x 4 N-tiles
// (4 accumulators, 4 WMMAs per K-step). B tile (32K x 64N) is staged in LDS
// transposed (ldsB[n][k], row stride 40 halves = 80B, 16B aligned) so each
// lane reads its 16 B-halves as two contiguous 16B ds loads.
// WMMA lane layouts per CDNA5 ISA 7.12.2 (16x16x32 f16):
//   A lane L: row=L&15 ; K = {v<4 ? 2v : 16+2(v-4)} + (L<16?0:8) + {0,1}
//   B lane L: col=L&15 ; K = {v<4 ? 2v : 16+2(v-4)} + (L<16?0:8) + {0,1}
//   C/D lane L, vgpr v: row = v + 8*(L>>4), col = L&15
// epi: 0=none, 1=+bias, 2=+bias,BN,ReLU
__global__ void __launch_bounds__(128)
wmma_gemm(const float* __restrict__ A, int lda,
          const f16_t* __restrict__ B, int ldb,
          float* __restrict__ C, int ldc,
          int M, int Nc, int K, int accum, int epi,
          const float* __restrict__ bias,
          const float* __restrict__ gamma, const float* __restrict__ beta,
          const float* __restrict__ mean,  const float* __restrict__ var){
    __shared__ f16_t ldsB[64][40];            // [n][k], padded row stride (80B)

    const int tid  = threadIdx.x;
    const int lane = tid & 31;
    const int wave = tid >> 5;
    const int rowW = blockIdx.x * 64 + wave * 16;   // this wave's 16 rows
    const int col0 = blockIdx.y * 64;               // block's 64 cols

    const int r     = lane & 15;
    const int hi    = lane >> 4;
    const int koffA = hi * 8;      // A lane-group K offset
    const int koffB = hi * 8;      // B lane-group K offset
    const bool rowOK = (rowW < M);                  // uniform per wave (M%16==0)
    const int arow  = min(rowW + r, M - 1);         // clamped load row

    // B stage assignment: thread t loads 16 contiguous halves of row k
    const int bK  = tid >> 2;          // 0..31
    const int bN0 = (tid & 3) * 16;    // 0,16,32,48

    v8f acc[4];
    #pragma unroll
    for (int nt = 0; nt < 4; ++nt){
        if (accum && rowOK){
            #pragma unroll
            for (int v = 0; v < 8; ++v)
                acc[nt][v] = C[(size_t)(rowW + v + 8 * hi) * ldc + col0 + nt * 16 + r];
        } else {
            acc[nt] = (v8f){0.f,0.f,0.f,0.f,0.f,0.f,0.f,0.f};
        }
    }

    const float* Ap = A + (size_t)arow * lda;
    for (int kk = 0; kk < K; kk += 32){
        // ---- stage B tile to LDS (transposed) ----
        const f16_t* Bg = B + (size_t)(kk + bK) * ldb + col0 + bN0;
        v8h b0 = *(const v8h*)(Bg);
        v8h b1 = *(const v8h*)(Bg + 8);
        #pragma unroll
        for (int j = 0; j < 8; ++j){
            ldsB[bN0 + j][bK]     = b0[j];
            ldsB[bN0 + 8 + j][bK] = b1[j];
        }
        __syncthreads();

        if (kk + 32 < K) __builtin_prefetch(Ap + kk + 32, 0, 0);  // global_prefetch_b8

        // ---- A fragment (f32 -> f16) ----
        v16h a;
        #pragma unroll
        for (int v = 0; v < 8; ++v){
            int kb = (v < 4 ? (v * 2) : (16 + (v - 4) * 2)) + koffA;
            float2 p = *(const float2*)(Ap + kk + kb);
            a[2 * v]     = (f16_t)p.x;
            a[2 * v + 1] = (f16_t)p.y;
        }

        // ---- 4 N-tiles: B from LDS, WMMA ----
        #pragma unroll
        for (int nt = 0; nt < 4; ++nt){
            const f16_t* bp = &ldsB[nt * 16 + r][koffB];
            v8h blo = *(const v8h*)(bp);
            v8h bhi = *(const v8h*)(bp + 16);
            v16h b;
            #pragma unroll
            for (int j = 0; j < 8; ++j){ b[j] = blo[j]; b[8 + j] = bhi[j]; }
            acc[nt] = __builtin_amdgcn_wmma_f32_16x16x32_f16(false, a, false, b,
                                                             (short)0, acc[nt], false, false);
        }
        __syncthreads();
    }

    if (!rowOK) return;
    #pragma unroll
    for (int nt = 0; nt < 4; ++nt){
        #pragma unroll
        for (int v = 0; v < 8; ++v){
            int rr = rowW + v + 8 * hi, cc = col0 + nt * 16 + r;
            float x = acc[nt][v];
            if (epi >= 1) x += bias[cc];
            if (epi == 2){
                x = (x - mean[cc]) * rsqrtf(var[cc] + 1e-5f) * gamma[cc] + beta[cc];
                x = fmaxf(x, 0.f);
            }
            C[(size_t)rr * ldc + cc] = x;
        }
    }
}

// ---------------- host orchestration ----------------
extern "C" void kernel_launch(void* const* d_in, const int* in_sizes, int n_in,
                              void* d_out, int out_size, void* d_ws, size_t ws_size,
                              hipStream_t stream){
    (void)in_sizes; (void)n_in; (void)out_size; (void)ws_size;
    const float* x     = (const float*)d_in[0];
    const int*   ei    = (const int*)d_in[1];
    const int*   src   = ei;            // edge_index[0]
    const int*   dst   = ei + EE;       // edge_index[1]
    const int*   perm1 = (const int*)d_in[2];
    const int*   perm2 = (const int*)d_in[3];
    // sage: [4..12] Wl,bl,Wr x3 ; bn: [13..20] ; gat: [21..56] (W,att_s,att_d,bias)x9
    const float* sWl[3] = {(const float*)d_in[4], (const float*)d_in[7], (const float*)d_in[10]};
    const float* sbl[3] = {(const float*)d_in[5], (const float*)d_in[8], (const float*)d_in[11]};
    const float* sWr[3] = {(const float*)d_in[6], (const float*)d_in[9], (const float*)d_in[12]};
    const float* bnG[2] = {(const float*)d_in[13], (const float*)d_in[17]};
    const float* bnB[2] = {(const float*)d_in[14], (const float*)d_in[18]};
    const float* bnM[2] = {(const float*)d_in[15], (const float*)d_in[19]};
    const float* bnV[2] = {(const float*)d_in[16], (const float*)d_in[20]};
    auto gp = [&](int i, int j, int k){ return (const float*)d_in[21 + (i * 3 + j) * 4 + k]; };
    const float* projW = (const float*)d_in[57];
    const float* projB = (const float*)d_in[58];
    float* out = (float*)d_out;

    // ---- workspace arena ----
    char* ws = (char*)d_ws;
    size_t cur = 0;
    auto alloc = [&](size_t bytes)->char*{
        char* p = ws + cur; cur += (bytes + 255) & ~(size_t)255; return p; };
    float*    P0      = (float*)alloc((size_t)NN * 512 * 4);
    float*    P1      = (float*)alloc((size_t)NN * 512 * 4);
    float*    g_x     = (float*)alloc((size_t)NN * 64 * 4);
    float*    g_x2    = (float*)alloc((size_t)NN * 64 * 4);
    float*    g_small = (float*)alloc((size_t)NN * 64 * 4);
    float*    f0      = (float*)alloc((size_t)NN * 64 * 4);
    float*    f1      = (float*)alloc((size_t)NN * 64 * 4);
    float*    f2      = (float*)alloc((size_t)NN * 64 * 4);
    float*    asrc    = (float*)alloc((size_t)NN * 8 * 4);
    float*    adst    = (float*)alloc((size_t)NN * 8 * 4);
    unsigned* nmax    = (unsigned*)alloc((size_t)NN * 8 * 4);
    float*    mfl     = (float*)alloc((size_t)NN * 8 * 4);
    float*    nden    = (float*)alloc((size_t)NN * 8 * 4);
    float*    edge_p  = (float*)alloc((size_t)EE * 8 * 4);
    float*    cnt     = (float*)alloc((size_t)NN * 4);
    float*    comb    = (float*)alloc((size_t)NN * 192 * 4);
    f16_t*    w16     = (f16_t*)alloc((size_t)1100000 * 2);

    const int TB = 256;
    auto zero = [&](float* p, long n){ k_zero<<<cdiv(n, TB), TB, 0, stream>>>(p, n); };
    auto pack = [&](const float* s, f16_t* d, long n){
        k_pack_f16<<<cdiv(n, TB), TB, 0, stream>>>(s, d, n); };
    auto gemm = [&](const float* A, int lda, const f16_t* B, int ldb, float* C, int ldc,
                    int M, int Nc, int K, int accum, int epi, const float* bias,
                    const float* g, const float* b, const float* m, const float* v){
        dim3 grid(cdiv(M, 64), Nc / 64);
        wmma_gemm<<<grid, 128, 0, stream>>>(A, lda, B, ldb, C, ldc, M, Nc, K,
                                            accum, epi, bias, g, b, m, v); };

    // ---- pack all weights to f16 ----
    f16_t* wSl[3]; f16_t* wSr[3]; f16_t* wG[3][3]; f16_t* wP;
    size_t hoff = 0;
    auto halloc = [&](size_t n)->f16_t*{ f16_t* p = w16 + hoff; hoff += n; return p; };
    wSl[0] = halloc(128 * 64); wSr[0] = halloc(128 * 64);
    wSl[1] = halloc(64 * 64);  wSr[1] = halloc(64 * 64);
    wSl[2] = halloc(64 * 64);  wSr[2] = halloc(64 * 64);
    for (int i = 0; i < 3; ++i){
        wG[i][0] = halloc(64 * 512);
        wG[i][1] = halloc(512 * 512);
        wG[i][2] = halloc(512 * 64);
    }
    wP = halloc(192 * 128);
    pack(sWl[0], wSl[0], 128 * 64); pack(sWr[0], wSr[0], 128 * 64);
    pack(sWl[1], wSl[1], 64 * 64);  pack(sWr[1], wSr[1], 64 * 64);
    pack(sWl[2], wSl[2], 64 * 64);  pack(sWr[2], wSr[2], 64 * 64);
    for (int i = 0; i < 3; ++i){
        pack(gp(i, 0, 0), wG[i][0], 64 * 512);
        pack(gp(i, 1, 0), wG[i][1], 512 * 512);
        pack(gp(i, 2, 0), wG[i][2], 512 * 64);
    }
    pack(projW, wP, 192 * 128);

    // ---- GraphSAGE encoder ----
    auto sage = [&](const float* xin, int Din, float* hout, int layer, int bn /* -1 none */){
        zero(P0, (long)NN * Din); zero(cnt, NN);
        k_edge_accum<<<cdiv((long)EE * Din, TB), TB, 0, stream>>>(xin, src, dst, P0, cnt, Din);
        k_mean<<<cdiv((long)NN * Din, TB), TB, 0, stream>>>(P0, cnt, NN, Din);
        gemm(P0, Din, wSl[layer], 64, hout, 64, NN, 64, Din, 0, 0,
             nullptr, nullptr, nullptr, nullptr, nullptr);
        if (bn >= 0)
            gemm(xin, Din, wSr[layer], 64, hout, 64, NN, 64, Din, 1, 2,
                 sbl[layer], bnG[bn], bnB[bn], bnM[bn], bnV[bn]);
        else
            gemm(xin, Din, wSr[layer], 64, hout, 64, NN, 64, Din, 1, 1,
                 sbl[layer], nullptr, nullptr, nullptr, nullptr);
    };
    sage(x,       128, g_small, 0, 0);   // h0 -> g_small
    sage(g_small, 64,  f0,      1, 1);   // h1 -> f0
    sage(f0,      64,  g_x,     2, -1);  // h2 -> g_x  (processor-0 input)

    // ---- GAT conv ----
    auto gatconv = [&](const float* xin, int Din, int n, int Hn, int pi, int lj,
                       float* T, float* O, int lim, int do_elu){
        int HC = Hn * 64;
        gemm(xin, Din, wG[pi][lj], HC, T, HC, n, HC, Din, 0, 0,
             nullptr, nullptr, nullptr, nullptr, nullptr);
        long nh = (long)n * Hn;
        k_att_node<<<cdiv(nh, TB), TB, 0, stream>>>(T, gp(pi, lj, 1), gp(pi, lj, 2),
                                                    asrc, adst, nmax, n, Hn);
        k_edge_max<<<cdiv((long)EE * Hn, TB), TB, 0, stream>>>(src, dst, asrc, adst,
                                                               nmax, Hn, lim);
        k_node_init<<<cdiv(nh, TB), TB, 0, stream>>>(T, asrc, adst, nmax, mfl, nden, O, n, Hn);
        k_edge_soft<<<cdiv((long)EE * Hn, TB), TB, 0, stream>>>(src, dst, asrc, adst,
                                                                mfl, nden, edge_p, Hn, lim);
        k_edge_scatter<<<cdiv((long)EE * Hn, TB), TB, 0, stream>>>(src, dst, edge_p,
                                                                   T, O, Hn, lim);
        k_gat_finish<<<cdiv((long)n * HC, TB), TB, 0, stream>>>(O, nden, gp(pi, lj, 3),
                                                                n, Hn, do_elu);
    };
    auto gatnet = [&](const float* xin, int n, int lim, int pi, float* fout){
        gatconv(xin, 64,  n, 8, pi, 0, P0, P1, lim, 1);       // elu(GAT 64->512)
        gatconv(P1,  512, n, 8, pi, 1, P0, P1, lim, 1);       // elu(GAT 512->512)
        gatconv(P1,  512, n, 1, pi, 2, g_small, fout, lim, 0);// GAT 512->64
    };

    gatnet(g_x, NN, NN, 0, f0);
    k_gather<<<cdiv((long)NN1 * 64, TB), TB, 0, stream>>>(g_x2, g_x, perm1, NN1, 64);
    gatnet(g_x2, NN1, NN1, 1, f1);
    k_gather<<<cdiv((long)NN2 * 64, TB), TB, 0, stream>>>(g_x, g_x2, perm2, NN2, 64);
    gatnet(g_x, NN2, NN2, 2, f2);

    // ---- pad + concat + projection ----
    k_combine<<<cdiv((long)NN * 192, TB), TB, 0, stream>>>(f0, f1, f2, comb);
    gemm(comb, 192, wP, 128, out, 128, NN, 128, 192, 0, 1,
         projB, nullptr, nullptr, nullptr, nullptr);
}
